// OrdinalIPAttnProcessor_30116310679947
// MI455X (gfx1250) — compile-verified, matrix operand order checked by reading, not run on hardware
//
#include <hip/hip_runtime.h>
#include <hip/hip_bf16.h>

typedef __attribute__((ext_vector_type(16))) _Float16 v16h;
typedef __attribute__((ext_vector_type(8)))  float    v8f;

#define WMMA16(A, B, C) \
  __builtin_amdgcn_wmma_f32_16x16x32_f16(false, (A), false, (B), (short)0, (C), false, false)

// ---- CDNA5 async global->LDS staging (guarded; falls back to sync copy) ----
// Probe-derived prototype: (AS1 v4i*, AS3 v4i*, imm offset, imm cpol)
typedef int v4i_vs __attribute__((vector_size(16)));
typedef __attribute__((address_space(1))) v4i_vs* g_v4ip;
typedef __attribute__((address_space(3))) v4i_vs* l_v4ip;
#if __has_builtin(__builtin_amdgcn_global_load_async_to_lds_b128)
#define HAVE_ASYNC_LDS 1
#define ASYNC_LDS_B128(g, l) \
  __builtin_amdgcn_global_load_async_to_lds_b128((g_v4ip)(g), (l_v4ip)(l), 0, 0)
#if __has_builtin(__builtin_amdgcn_s_wait_asynccnt)
#define WAIT_ASYNC() __builtin_amdgcn_s_wait_asynccnt(0)
#else
#define WAIT_ASYNC() asm volatile("s_wait_asynccnt 0" ::: "memory")
#endif
#else
#define HAVE_ASYNC_LDS 0
#endif

namespace {
constexpr int Bc = 4, Sc = 4096, Cc = 1280, Hc = 20, Dc = 64;
constexpr int CROSS = 768, AOEc = 77, IPc = 4;
constexpr int Mc = Bc * Sc;          // 16384 rows
constexpr int KP  = 80;              // AOE keys padded for N-tiling (scores)
constexpr int KP2 = 96;              // AOE keys padded for K-tiling (P @ V)
constexpr int IPN = 16;              // IP keys padded (N)
constexpr int IPK = 32;              // IP keys padded (K)
constexpr float SCALEc = 0.125f;     // D^-0.5
}

template <bool F16> struct AElemT { using T = float; };
template <> struct AElemT<true>   { using T = _Float16; };

// k-index of the low element of the f16 pair held in fragment VGPR v, per ISA layout
__device__ __forceinline__ int frag_k(int lane, int v) {
  return ((v & 4) << 2) + ((lane & 16) >> 1) + ((v & 3) << 1);
}

// ---------------------------------------------------------------------------
// KV projections: tiny GEMVs, one block per output row of one projection.
// K stored [b][h][key][d]; V stored transposed [b][h][d][key] (padded).
// ---------------------------------------------------------------------------
__global__ __launch_bounds__(256)
void kv_proj_kernel(const float* __restrict__ enc,
                    const float* __restrict__ Wk,  const float* __restrict__ Wv,
                    const float* __restrict__ Wkip, const float* __restrict__ Wvip,
                    _Float16* __restrict__ kbuf, _Float16* __restrict__ vbuf,
                    _Float16* __restrict__ kipb, _Float16* __restrict__ vipb)
{
  const int blk = blockIdx.x;
  int job, r;
  if      (blk <     Bc * AOEc)            { job = 0; r = blk; }
  else if (blk < 2 * Bc * AOEc)            { job = 1; r = blk - Bc * AOEc; }
  else if (blk < 2 * Bc * AOEc + Bc * IPc) { job = 2; r = blk - 2 * Bc * AOEc; }
  else                                     { job = 3; r = blk - 2 * Bc * AOEc - Bc * IPc; }
  const bool isV  = (job & 1);
  const bool isIP = (job >= 2);
  const int  tokN = isIP ? IPc : AOEc;
  const int  b    = r / tokN;
  const int  key  = r % tokN;
  const float* W  = (job == 0) ? Wk : (job == 1) ? Wv : (job == 2) ? Wkip : Wvip;
  const float* tok = enc + ((size_t)b * (AOEc + IPc) + (isIP ? AOEc + key : key)) * CROSS;

  const int t = threadIdx.x;
  float acc[5] = {0.f, 0.f, 0.f, 0.f, 0.f};
  for (int k = 0; k < CROSS; ++k) {
    const float x = tok[k];                       // uniform -> scalar load
    #pragma unroll
    for (int c = 0; c < 5; ++c)
      acc[c] = fmaf(x, W[(size_t)k * Cc + t + c * 256], acc[c]);
  }
  #pragma unroll
  for (int c = 0; c < 5; ++c) {
    const int col = t + c * 256;
    const int h = col >> 6, d = col & 63;
    const size_t bh = (size_t)b * Hc + h;
    const _Float16 val = (_Float16)acc[c];
    if (!isIP) {
      if (!isV) kbuf[(bh * KP  + key) * Dc  + d  ] = val;
      else      vbuf[(bh * Dc  + d  ) * KP2 + key] = val;
    } else {
      if (!isV) kipb[(bh * IPN + key) * Dc  + d  ] = val;
      else      vipb[(bh * Dc  + d  ) * IPK + key] = val;
    }
  }
}

// ---------------------------------------------------------------------------
// Tiled WMMA GEMM: C[M x 1280] = A[M x 1280] * B[1280 x 1280] (+bias+resid)
// Block tile 64x128, K-step 64, 8 waves -> each wave does 8 WMMAs/iteration.
// A tile staged RAW via async global->LDS (f32->f16 converted at frag build);
// B tile staged sync with convert + transpose.
// ---------------------------------------------------------------------------
template <bool A_F16, bool FUSE_OUT>
__global__ __launch_bounds__(256)
void wmma_gemm_kernel(const void* __restrict__ Ap,
                      const float* __restrict__ Bw,
                      const float* __restrict__ bias,
                      const float* __restrict__ resid,
                      void* __restrict__ Outp)
{
  using AT = typename AElemT<A_F16>::T;
  constexpr int MB = 64, NB = 128, KS = 64;
  // element strides chosen so 16 fragment lanes hit 16 distinct LDS banks,
  // and rows stay 16B-aligned for b128 async writes
  constexpr int LDA = A_F16 ? 72 : 68;
  constexpr int LDB = 66;
  __shared__ __align__(16) AT       As[MB * LDA];
  __shared__ __align__(16) _Float16 Bs[NB * LDB];   // transposed: Bs[n*LDB + k]

  const int t    = threadIdx.x;
  const int lane = t & 31;
  const int w    = t >> 5;
  const int wm   = w >> 2;       // 0..1
  const int wn   = w & 3;        // 0..3
  const int m0   = blockIdx.x * MB;
  const int n0   = blockIdx.y * NB;
  const int lm   = lane & 15;

  const AT* __restrict__ A = (const AT*)Ap;

  v8f acc[2][2] = {};

  for (int k0 = 0; k0 < Cc; k0 += KS) {
    // ---- stage A tile (64 x 64) raw ----
#if HAVE_ASYNC_LDS
    {
      constexpr int EPO = 16 / (int)sizeof(AT);        // elements per b128 op
      constexpr int OPS = (MB * KS) / EPO / 256;       // ops per thread
      #pragma unroll
      for (int i = 0; i < OPS; ++i) {
        const int o = t + i * 256;
        const int r = o / (KS / EPO);
        const int c = (o % (KS / EPO)) * EPO;
        ASYNC_LDS_B128(A + (size_t)(m0 + r) * Cc + k0 + c, &As[r * LDA + c]);
      }
    }
#else
    #pragma unroll
    for (int i = 0; i < (MB * KS) / 256; ++i) {
      const int e = t + i * 256;
      const int r = e >> 6, c = e & 63;
      As[r * LDA + c] = A[(size_t)(m0 + r) * Cc + k0 + c];
    }
#endif
    // ---- stage B tile (64 x 128) -> f16, transposed ----
    #pragma unroll
    for (int i = 0; i < (KS * NB) / 256; ++i) {
      const int e = t + i * 256;
      const int r = e >> 7, c = e & 127;
      Bs[c * LDB + r] = (_Float16)Bw[(size_t)(k0 + r) * Cc + n0 + c];
    }
#if HAVE_ASYNC_LDS
    WAIT_ASYNC();
#endif
    __syncthreads();

    v16h af[2][2], bf[2][2];
    #pragma unroll
    for (int mt = 0; mt < 2; ++mt) {
      const int m = wm * 32 + mt * 16 + lm;
      #pragma unroll
      for (int kk = 0; kk < 2; ++kk)
        #pragma unroll
        for (int v = 0; v < 8; ++v) {
          const int k = kk * 32 + frag_k(lane, v);
          af[mt][kk][2 * v]     = (_Float16)As[m * LDA + k];
          af[mt][kk][2 * v + 1] = (_Float16)As[m * LDA + k + 1];
        }
    }
    #pragma unroll
    for (int nt = 0; nt < 2; ++nt) {
      const int n = wn * 32 + nt * 16 + lm;
      #pragma unroll
      for (int kk = 0; kk < 2; ++kk)
        #pragma unroll
        for (int v = 0; v < 8; ++v) {
          const int k = kk * 32 + frag_k(lane, v);
          bf[nt][kk][2 * v]     = Bs[n * LDB + k];
          bf[nt][kk][2 * v + 1] = Bs[n * LDB + k + 1];
        }
    }
    #pragma unroll
    for (int mt = 0; mt < 2; ++mt)
      #pragma unroll
      for (int nt = 0; nt < 2; ++nt) {
        acc[mt][nt] = WMMA16(af[mt][0], bf[nt][0], acc[mt][nt]);
        acc[mt][nt] = WMMA16(af[mt][1], bf[nt][1], acc[mt][nt]);
      }
    __syncthreads();
  }

  const int rowoff = (lane & 16) ? 8 : 0;
  #pragma unroll
  for (int mt = 0; mt < 2; ++mt)
    #pragma unroll
    for (int nt = 0; nt < 2; ++nt)
      #pragma unroll
      for (int r = 0; r < 8; ++r) {
        const int row = m0 + wm * 32 + mt * 16 + r + rowoff;
        const int col = n0 + wn * 32 + nt * 16 + lm;
        const float val = acc[mt][nt][r];
        if constexpr (FUSE_OUT) {
          ((float*)Outp)[(size_t)row * Cc + col] =
              val + bias[col] + resid[(size_t)row * Cc + col];
        } else {
          ((_Float16*)Outp)[(size_t)row * Cc + col] = (_Float16)val;
        }
      }
}

// ---------------------------------------------------------------------------
// Dual cross-attention: one wave per (b, h, 16-query tile). WMMA scores,
// masked softmax through LDS, WMMA P@V; IP path accumulated into same out.
// Writes result in place over the consumed Q tile.
// ---------------------------------------------------------------------------
__global__ __launch_bounds__(128)
void attn_kernel(_Float16* __restrict__ qattn,
                 const _Float16* __restrict__ kbuf,
                 const _Float16* __restrict__ vbuf,
                 const _Float16* __restrict__ kipb,
                 const _Float16* __restrict__ vipb)
{
  __shared__ float    sS[4][16][KP];
  __shared__ _Float16 sP[4][16][KP2];
  __shared__ _Float16 sPip[4][16][IPK];

  const int t = threadIdx.x, lane = t & 31, w = t >> 5;
  const int gw = blockIdx.x * 4 + w;
  const int qt = gw & 255;           // S/16 = 256 query tiles
  const int bh = gw >> 8;
  const int h  = bh % Hc;
  const int b  = bh / Hc;
  const int q0 = qt * 16;
  const int lm = lane & 15;
  const int rowoff = (lane & 16) ? 8 : 0;

  _Float16* __restrict__ Qb = qattn + ((size_t)b * Sc + q0) * Cc + h * Dc;

  // Q tile as A-fragments (16x64 -> two 16x32 frags)
  v16h aq[2];
  #pragma unroll
  for (int kk = 0; kk < 2; ++kk)
    #pragma unroll
    for (int v = 0; v < 8; ++v) {
      const int k = kk * 32 + frag_k(lane, v);
      aq[kk][2 * v]     = Qb[(size_t)lm * Cc + k];
      aq[kk][2 * v + 1] = Qb[(size_t)lm * Cc + k + 1];
    }

  // ---- AOE scores: 16 x 80 ----
  const _Float16* Kb = kbuf + (size_t)(b * Hc + h) * KP * Dc;
  #pragma unroll
  for (int nt = 0; nt < 5; ++nt) {
    v16h b0, b1;
    const int key = nt * 16 + lm;
    #pragma unroll
    for (int v = 0; v < 8; ++v) {
      const int k = frag_k(lane, v);
      b0[2 * v]     = Kb[key * Dc + k];
      b0[2 * v + 1] = Kb[key * Dc + k + 1];
      b1[2 * v]     = Kb[key * Dc + 32 + k];
      b1[2 * v + 1] = Kb[key * Dc + 32 + k + 1];
    }
    v8f sc = {};
    sc = WMMA16(aq[0], b0, sc);
    sc = WMMA16(aq[1], b1, sc);
    #pragma unroll
    for (int r = 0; r < 8; ++r)
      sS[w][r + rowoff][nt * 16 + lm] = sc[r];
  }
  __syncthreads();

  // ---- AOE softmax (rows owned by lanes 0..15) ----
  if (lane < 16) {
    const int row = lane;
    float mx = -1e30f;
    for (int c = 0; c < AOEc; ++c) {
      const float s = sS[w][row][c] * SCALEc;
      sS[w][row][c] = s;
      mx = fmaxf(mx, s);
    }
    float sum = 0.f;
    for (int c = 0; c < AOEc; ++c) {
      const float e = __expf(sS[w][row][c] - mx);
      sS[w][row][c] = e;
      sum += e;
    }
    const float inv = 1.f / sum;
    for (int c = 0; c < KP2; ++c)
      sP[w][row][c] = (c < AOEc) ? (_Float16)(sS[w][row][c] * inv) : (_Float16)0.f;
  }
  __syncthreads();

  // ---- out += P_aoe @ V_aoe  (K padded to 96) ----
  v8f o[4] = {};
  const _Float16* Vb = vbuf + (size_t)(b * Hc + h) * Dc * KP2;
  #pragma unroll
  for (int kk = 0; kk < 3; ++kk) {
    v16h ap;
    #pragma unroll
    for (int v = 0; v < 8; ++v) {
      const int k = kk * 32 + frag_k(lane, v);
      ap[2 * v]     = sP[w][lm][k];
      ap[2 * v + 1] = sP[w][lm][k + 1];
    }
    #pragma unroll
    for (int nt = 0; nt < 4; ++nt) {
      v16h bv;
      const int d = nt * 16 + lm;
      #pragma unroll
      for (int v = 0; v < 8; ++v) {
        const int k = kk * 32 + frag_k(lane, v);
        bv[2 * v]     = Vb[d * KP2 + k];
        bv[2 * v + 1] = Vb[d * KP2 + k + 1];
      }
      o[nt] = WMMA16(ap, bv, o[nt]);
    }
  }

  // ---- IP scores: 16 x 16 (4 real keys) ----
  const _Float16* Kip = kipb + (size_t)(b * Hc + h) * IPN * Dc;
  v16h bi0, bi1;
  #pragma unroll
  for (int v = 0; v < 8; ++v) {
    const int k = frag_k(lane, v);
    bi0[2 * v]     = Kip[lm * Dc + k];
    bi0[2 * v + 1] = Kip[lm * Dc + k + 1];
    bi1[2 * v]     = Kip[lm * Dc + 32 + k];
    bi1[2 * v + 1] = Kip[lm * Dc + 32 + k + 1];
  }
  v8f sip = {};
  sip = WMMA16(aq[0], bi0, sip);
  sip = WMMA16(aq[1], bi1, sip);
  #pragma unroll
  for (int r = 0; r < 8; ++r)
    sS[w][r + rowoff][lm] = sip[r];
  __syncthreads();

  if (lane < 16) {
    const int row = lane;
    float mx = -1e30f;
    for (int c = 0; c < IPc; ++c) {
      const float s = sS[w][row][c] * SCALEc;
      sS[w][row][c] = s;
      mx = fmaxf(mx, s);
    }
    float sum = 0.f;
    for (int c = 0; c < IPc; ++c) {
      const float e = __expf(sS[w][row][c] - mx);
      sS[w][row][c] = e;
      sum += e;
    }
    const float inv = 1.f / sum;
    for (int c = 0; c < IPK; ++c)
      sPip[w][row][c] = (c < IPc) ? (_Float16)(sS[w][row][c] * inv) : (_Float16)0.f;
  }
  __syncthreads();

  // ---- out += P_ip @ V_ip (all scales == 1.0) ----
  const _Float16* Vip = vipb + (size_t)(b * Hc + h) * Dc * IPK;
  v16h api;
  #pragma unroll
  for (int v = 0; v < 8; ++v) {
    const int k = frag_k(lane, v);
    api[2 * v]     = sPip[w][lm][k];
    api[2 * v + 1] = sPip[w][lm][k + 1];
  }
  #pragma unroll
  for (int nt = 0; nt < 4; ++nt) {
    v16h bv;
    const int d = nt * 16 + lm;
    #pragma unroll
    for (int v = 0; v < 8; ++v) {
      const int k = frag_k(lane, v);
      bv[2 * v]     = Vip[d * IPK + k];
      bv[2 * v + 1] = Vip[d * IPK + k + 1];
    }
    o[nt] = WMMA16(api, bv, o[nt]);
  }

  // ---- in-place store over the Q tile this wave consumed ----
  #pragma unroll
  for (int nt = 0; nt < 4; ++nt)
    #pragma unroll
    for (int r = 0; r < 8; ++r)
      Qb[(size_t)(r + rowoff) * Cc + nt * 16 + lm] = (_Float16)o[nt][r];
}

// ---------------------------------------------------------------------------
extern "C" void kernel_launch(void* const* d_in, const int* in_sizes, int n_in,
                              void* d_out, int out_size, void* d_ws, size_t ws_size,
                              hipStream_t stream)
{
  (void)in_sizes; (void)n_in; (void)out_size; (void)ws_size;
  const float* hidden = (const float*)d_in[0];
  const float* enc    = (const float*)d_in[1];
  const float* Wq     = (const float*)d_in[2];
  const float* Wk     = (const float*)d_in[3];
  const float* Wv     = (const float*)d_in[4];
  const float* Wkip   = (const float*)d_in[5];
  const float* Wvip   = (const float*)d_in[6];
  const float* Wo     = (const float*)d_in[7];
  const float* bo     = (const float*)d_in[8];
  float* out = (float*)d_out;

  char* ws = (char*)d_ws;
  _Float16* qattn = (_Float16*)ws;
  size_t off = (size_t)Mc * Cc * sizeof(_Float16);
  _Float16* kbuf = (_Float16*)(ws + off); off += (size_t)Bc * Hc * KP  * Dc  * 2;
  _Float16* vbuf = (_Float16*)(ws + off); off += (size_t)Bc * Hc * Dc  * KP2 * 2;
  _Float16* kipb = (_Float16*)(ws + off); off += (size_t)Bc * Hc * IPN * Dc  * 2;
  _Float16* vipb = (_Float16*)(ws + off); off += (size_t)Bc * Hc * Dc  * IPK * 2;
  const size_t kvbytes = off - (size_t)Mc * Cc * 2;

  // zero KV pads (graph-capture safe)
  (void)hipMemsetAsync(kbuf, 0, kvbytes, stream);

  // KV projections (308 + 308 + 16 + 16 rows)
  kv_proj_kernel<<<2 * Bc * AOEc + 2 * Bc * IPc, 256, 0, stream>>>(
      enc, Wk, Wv, Wkip, Wvip, kbuf, vbuf, kipb, vipb);

  // Q projection: f32 in -> f16 out
  dim3 ggrid(Mc / 64, Cc / 128);
  wmma_gemm_kernel<false, false><<<ggrid, 256, 0, stream>>>(
      hidden, Wq, nullptr, nullptr, qattn);

  // dual cross-attention, in place over qattn
  attn_kernel<<<(Bc * Hc * (Sc / 16)) / 4, 128, 0, stream>>>(
      qattn, kbuf, vbuf, kipb, vipb);

  // O projection with fused bias + residual, fp32 out
  wmma_gemm_kernel<true, true><<<ggrid, 256, 0, stream>>>(
      qattn, Wo, bo, hidden, out);
}
